// GPTTransformerBlock_26199300505993
// MI455X (gfx1250) — matrix operand order repaired and checked by reading, not compiled, not run on hardware
//
#include <hip/hip_runtime.h>
#include <math.h>

typedef __attribute__((ext_vector_type(16))) __bf16        v16bf;
typedef __attribute__((ext_vector_type(8)))  float         v8f;
typedef __attribute__((ext_vector_type(4)))  unsigned int  u32x4;
typedef int v4i __attribute__((vector_size(16)));   // matches builtin param

union FragU { v16bf v; u32x4 u[2]; };
union Pack8 { u32x4 u; __bf16 h[8]; };

// ---------------------------------------------------------------------------
// Async global->LDS copy (CDNA5 GLOBAL_LOAD_ASYNC_TO_LDS_B128, ASYNCcnt).
// Guarded: falls back to load+store staging if the builtin is absent.
// ---------------------------------------------------------------------------
#if defined(__has_builtin)
#if __has_builtin(__builtin_amdgcn_global_load_async_to_lds_b128)
#define HAVE_ASYNC_LDS 1
#endif
#endif

__device__ __forceinline__ void copy16_to_lds(__bf16* lds, const __bf16* g) {
#ifdef HAVE_ASYNC_LDS
  __builtin_amdgcn_global_load_async_to_lds_b128(
      (__attribute__((address_space(1))) v4i*)g,
      (__attribute__((address_space(3))) v4i*)lds,
      /*offset=*/0, /*cpol=*/0);
#else
  *(u32x4*)lds = *(const u32x4*)g;
#endif
}

__device__ __forceinline__ void wait_async_lds() {
#ifdef HAVE_ASYNC_LDS
#if __has_builtin(__builtin_amdgcn_s_wait_asynccnt)
  __builtin_amdgcn_s_wait_asynccnt(0);
#else
  asm volatile("s_wait_asynccnt 0x0" ::: "memory");
#endif
#endif
}

// ---------------------------------------------------------------------------
// WMMA wrapper: D = A(16x32 bf16) * B(32x16 bf16) + C(16x16 f32)
// ---------------------------------------------------------------------------
__device__ __forceinline__ v8f wmma_bf16(v16bf a, v16bf b, v8f c) {
  return __builtin_amdgcn_wmma_f32_16x16x32_bf16(
      /*neg_a=*/false, a, /*neg_b=*/false, b,
      /*c_mod=*/(short)0, c, /*reuse_a=*/false, /*reuse_b=*/false);
}

// ---------------------------------------------------------------------------
// DPP row_xmask butterfly reductions within each 16-lane row (pure VALU).
// ---------------------------------------------------------------------------
template <int MASK>
__device__ __forceinline__ float dpp_xor16(float x) {
  int xi = __float_as_int(x);
  int yi = __builtin_amdgcn_update_dpp(xi, xi, 0x160 | MASK, 0xF, 0xF, true);
  return __int_as_float(yi);
}
__device__ __forceinline__ float rowmax16(float x) {
  x = fmaxf(x, dpp_xor16<1>(x));
  x = fmaxf(x, dpp_xor16<2>(x));
  x = fmaxf(x, dpp_xor16<4>(x));
  x = fmaxf(x, dpp_xor16<8>(x));
  return x;
}
__device__ __forceinline__ float rowsum16(float x) {
  x += dpp_xor16<1>(x);
  x += dpp_xor16<2>(x);
  x += dpp_xor16<4>(x);
  x += dpp_xor16<8>(x);
  return x;
}

// ---------------------------------------------------------------------------
// Fragment loaders (wave32 ISA layouts), all via 16-byte vector loads.
// ---------------------------------------------------------------------------
__device__ __forceinline__ v16bf frag_a(const __bf16* base, int ld) {
  int lane = threadIdx.x & 31;
  const __bf16* p = base + (size_t)(lane & 15) * ld + (((lane >> 4) & 1) << 3);
  FragU f;
  f.u[0] = *(const u32x4*)p;
  f.u[1] = *(const u32x4*)(p + 16);
  return f.v;
}

// B frag 32x16 (KxN) from a TRANSPOSED tile stored [N][K]:
__device__ __forceinline__ v16bf frag_bT(const __bf16* base, int ld) {
  int lane = threadIdx.x & 31;
  const __bf16* p = base + (size_t)(lane & 15) * ld + (((lane >> 4) & 1) << 3);
  FragU f;
  f.u[0] = *(const u32x4*)p;
  f.u[1] = *(const u32x4*)(p + 16);
  return f.v;
}

// ---------------------------------------------------------------------------
// One-time weight convert+transpose: W[K,N] fp32 -> Wt[N,K] bf16.
// ---------------------------------------------------------------------------
__global__ __launch_bounds__(256) void cvt_transpose_kernel(
    const float* __restrict__ W, __bf16* __restrict__ Wt, int K, int N) {
  int idx = blockIdx.x * 256 + threadIdx.x;
  int k = idx / N;
  int n = idx - k * N;
  Wt[(size_t)n * K + k] = (__bf16)W[idx];
}

// ---------------------------------------------------------------------------
// LayerNorm: one block per token row (D = 1024), 256 threads, bf16 out.
// ---------------------------------------------------------------------------
__global__ __launch_bounds__(256) void layernorm_kernel(
    const float* __restrict__ x, const float* __restrict__ g,
    const float* __restrict__ b, __bf16* __restrict__ y) {
  constexpr int D = 1024;
  const int row = blockIdx.x;
  const int tid = threadIdx.x;
  const float* xr = x + (size_t)row * D;

  __shared__ float red[8];

  float v[4];
  float s = 0.f;
#pragma unroll
  for (int i = 0; i < 4; ++i) {
    v[i] = xr[tid + i * 256];
    s += v[i];
  }
#pragma unroll
  for (int off = 16; off >= 1; off >>= 1) s += __shfl_xor(s, off, 32);
  if ((tid & 31) == 0) red[tid >> 5] = s;
  __syncthreads();
  float tot = 0.f;
#pragma unroll
  for (int i = 0; i < 8; ++i) tot += red[i];
  const float mean = tot * (1.0f / D);
  __syncthreads();

  float vs = 0.f;
#pragma unroll
  for (int i = 0; i < 4; ++i) {
    float d = v[i] - mean;
    vs += d * d;
  }
#pragma unroll
  for (int off = 16; off >= 1; off >>= 1) vs += __shfl_xor(vs, off, 32);
  if ((tid & 31) == 0) red[tid >> 5] = vs;
  __syncthreads();
  tot = 0.f;
#pragma unroll
  for (int i = 0; i < 8; ++i) tot += red[i];
  const float rstd = rsqrtf(tot * (1.0f / D) + 1e-5f);

  __bf16* yr = y + (size_t)row * D;
#pragma unroll
  for (int i = 0; i < 4; ++i) {
    int c = tid + i * 256;
    yr[c] = (__bf16)((v[i] - mean) * rstd * g[c] + b[c]);
  }
}

// ---------------------------------------------------------------------------
// bf16-WMMA GEMM:  C[M,N] = act(A[M,K] @ W[K,N] + bias) (+ res)
//   A:  bf16 [M,K] row-major ;  Wt: bf16 [N,K] (pre-transposed weights)
// Block tile 128x128, BK=64, double-buffered LDS (2 x 32 KB), async staging.
// 8 waves as 4(m) x 2(n); wave tile 32x64 -> 16 WMMAs per k-tile per wave.
// ---------------------------------------------------------------------------
template <bool RELU, bool HASRES, typename OutT>
__global__ __launch_bounds__(256) void gemm_bf16_kernel(
    const __bf16* __restrict__ A, const __bf16* __restrict__ Wt,
    const float* __restrict__ bias, const float* __restrict__ res,
    OutT* __restrict__ C, int M, int N, int K) {
  __shared__ __bf16 sA[2][128 * 64];   // [m][k]  2 x 16 KB
  __shared__ __bf16 sBt[2][128 * 64];  // [n][k]  2 x 16 KB

  const int tid = threadIdx.x;
  const int lane = tid & 31;
  const int wave = tid >> 5;
  const int wm = wave >> 1;  // 0..3  (rows, 32 each)
  const int wn = wave & 1;   // 0..1  (cols, 64 each)
  const int bm = blockIdx.y * 128;
  const int bn = blockIdx.x * 128;

  v8f acc[2][4];
#pragma unroll
  for (int mi = 0; mi < 2; ++mi)
#pragma unroll
    for (int nt = 0; nt < 4; ++nt) acc[mi][nt] = {};

  // staging coords: each thread moves 4 x 16B per tile (contiguous 64B)
  const int gr = tid >> 1;         // row 0..127
  const int gc = (tid & 1) * 32;   // col 0 or 32

  auto stage = [&](int buf, int kt) {
#pragma unroll
    for (int i = 0; i < 4; ++i) {
      copy16_to_lds(&sA[buf][(size_t)gr * 64 + gc + i * 8],
                    A + (size_t)(bm + gr) * K + kt + gc + i * 8);
      copy16_to_lds(&sBt[buf][(size_t)gr * 64 + gc + i * 8],
                    Wt + (size_t)(bn + gr) * K + kt + gc + i * 8);
    }
  };

  stage(0, 0);
  int cur = 0;
  for (int kt = 0; kt < K; kt += 64) {
    wait_async_lds();     // my staged tile is in LDS
    __syncthreads();      // everyone's staged tile is in LDS
    if (kt + 64 < K) stage(cur ^ 1, kt + 64);  // overlap with WMMAs below
#pragma unroll
    for (int ks = 0; ks < 64; ks += 32) {
      v16bf a0 = frag_a(&sA[cur][(size_t)(wm * 32 + 0) * 64 + ks], 64);
      v16bf a1 = frag_a(&sA[cur][(size_t)(wm * 32 + 16) * 64 + ks], 64);
#pragma unroll
      for (int nt = 0; nt < 4; ++nt) {
        v16bf b = frag_bT(&sBt[cur][(size_t)(wn * 64 + nt * 16) * 64 + ks], 64);
        acc[0][nt] = wmma_bf16(a0, b, acc[0][nt]);
        acc[1][nt] = wmma_bf16(a1, b, acc[1][nt]);
      }
    }
    cur ^= 1;
  }

  const int rowBase = bm + wm * 32;
  const int colBase = bn + wn * 64;
#pragma unroll
  for (int mi = 0; mi < 2; ++mi)
#pragma unroll
    for (int nt = 0; nt < 4; ++nt)
#pragma unroll
      for (int i = 0; i < 8; ++i) {
        int row = rowBase + mi * 16 + (((lane >> 4) & 1) << 3) + i;
        int col = colBase + nt * 16 + (lane & 15);
        float vv = acc[mi][nt][i] + bias[col];
        if (RELU) vv = fmaxf(vv, 0.0f);
        if (HASRES) vv += res[(size_t)row * N + col];
        C[(size_t)row * N + col] = (OutT)vv;
      }
}

// ---------------------------------------------------------------------------
// Flash attention (causal), d_k = 64, bf16 Q/K/V/O.
// Block = 128 query rows of one (b,h); 8 waves own 16 rows each.
// K tile staged via async global->LDS; V staged transposed manually;
// softmax reductions are DPP row_xmask butterflies.
// ---------------------------------------------------------------------------
__global__ __launch_bounds__(256) void flash_attn_kernel(
    const __bf16* __restrict__ Q, const __bf16* __restrict__ K,
    const __bf16* __restrict__ V, __bf16* __restrict__ O,
    int H, int S, int Dm) {
  __shared__ __bf16 sK[32 * 64];      // [key][dk]   4 KB
  __shared__ __bf16 sVt[64 * 32];     // [dk][key]   4 KB
  __shared__ __bf16 sP[8][16 * 32];   // per-wave P  8 KB

  const int tid = threadIdx.x;
  const int lane = tid & 31;
  const int wave = tid >> 5;
  const int bh = blockIdx.x;
  const int bi = bh / H;
  const int hi = bh % H;
  const int q0 = blockIdx.y * 128;
  const int qr = q0 + wave * 16;
  const size_t hoff = (size_t)bi * S * Dm + (size_t)hi * 64;

  const __bf16* qbase = Q + hoff + (size_t)qr * Dm;
  v16bf qa0 = frag_a(qbase + 0, Dm);
  v16bf qa1 = frag_a(qbase + 32, Dm);

  float m[8], l[8];
  v8f o[4];
#pragma unroll
  for (int i = 0; i < 8; ++i) { m[i] = -1e30f; l[i] = 0.f; }
#pragma unroll
  for (int nt = 0; nt < 4; ++nt) o[nt] = {};

  const float scale = 0.125f;  // 1/sqrt(64)
  __bf16* pw = sP[wave];

  const int e0 = tid * 8;
  const int sr = e0 >> 6;   // key row 0..31
  const int sc = e0 & 63;   // dk col

  for (int kb0 = 0; kb0 < q0 + 128; kb0 += 32) {
    // K: async 16B copy to LDS.  V: 16B load, scatter-store transposed.
    copy16_to_lds(sK + e0, K + hoff + (size_t)(kb0 + sr) * Dm + sc);
    Pack8 t;
    t.u = *(const u32x4*)(V + hoff + (size_t)(kb0 + sr) * Dm + sc);
#pragma unroll
    for (int j = 0; j < 8; ++j) sVt[(size_t)(sc + j) * 32 + sr] = t.h[j];
    wait_async_lds();
    __syncthreads();

    // S = Q @ K^T : two 16x16 C frags (keys kb0..+15, kb0+16..+31)
    v8f s0 = {}, s1 = {};
    s0 = wmma_bf16(qa0, frag_bT(sK + 0 * 64 + 0, 64), s0);
    s0 = wmma_bf16(qa1, frag_bT(sK + 0 * 64 + 32, 64), s0);
    s1 = wmma_bf16(qa0, frag_bT(sK + 16 * 64 + 0, 64), s1);
    s1 = wmma_bf16(qa1, frag_bT(sK + 16 * 64 + 32, 64), s1);

    // Online softmax.  C frag: col = lane%16, rows = (lane/16)*8 + i.
#pragma unroll
    for (int i = 0; i < 8; ++i) {
      int riw = (((lane >> 4) & 1) << 3) + i;
      int rg = qr + riw;
      int c0 = kb0 + (lane & 15);
      int c1 = c0 + 16;
      float v0 = (c0 <= rg) ? s0[i] * scale : -1e30f;
      float v1 = (c1 <= rg) ? s1[i] * scale : -1e30f;
      float mx = rowmax16(fmaxf(v0, v1));
      float mn = fmaxf(m[i], mx);
      float alpha = __expf(m[i] - mn);
      float p0 = __expf(v0 - mn);
      float p1 = __expf(v1 - mn);
      float rs = rowsum16(p0 + p1);
      l[i] = l[i] * alpha + rs;
      m[i] = mn;
      o[0][i] *= alpha;
      o[1][i] *= alpha;
      o[2][i] *= alpha;
      o[3][i] *= alpha;
      pw[riw * 32 + (lane & 15)] = (__bf16)p0;
      pw[riw * 32 + 16 + (lane & 15)] = (__bf16)p1;
    }

    // O += P(16x32) @ V(32x64); B frags from transposed V tile.
    v16bf pa = frag_a(pw, 32);
#pragma unroll
    for (int nt = 0; nt < 4; ++nt)
      o[nt] = wmma_bf16(pa, frag_bT(sVt + (size_t)(nt * 16) * 32, 32), o[nt]);
    __syncthreads();
  }

  // Normalize, write context (bf16) in [B,S,D] layout.
#pragma unroll
  for (int i = 0; i < 8; ++i) {
    int rg = qr + (((lane >> 4) & 1) << 3) + i;
    float inv = 1.0f / l[i];
#pragma unroll
    for (int nt = 0; nt < 4; ++nt)
      O[hoff + (size_t)rg * Dm + nt * 16 + (lane & 15)] =
          (__bf16)(o[nt][i] * inv);
  }
}

// ---------------------------------------------------------------------------
// Launch: weight cvt+transpose -> LN1 -> Q/K/V -> flash -> O-proj(+res) ->
//         LN2 -> FFN1(ReLU) -> FFN2(+res -> d_out)
// ---------------------------------------------------------------------------
extern "C" void kernel_launch(void* const* d_in, const int* in_sizes, int n_in,
                              void* d_out, int out_size, void* d_ws,
                              size_t ws_size, hipStream_t stream) {
  (void)in_sizes; (void)n_in; (void)out_size; (void)ws_size;

  const float* x    = (const float*)d_in[0];
  const float* ln1g = (const float*)d_in[1];
  const float* ln1b = (const float*)d_in[2];
  const float* ln2g = (const float*)d_in[3];
  const float* ln2b = (const float*)d_in[4];
  const float* wq = (const float*)d_in[5];
  const float* bq = (const float*)d_in[6];
  const float* wk = (const float*)d_in[7];
  const float* bk = (const float*)d_in[8];
  const float* wv = (const float*)d_in[9];
  const float* bv = (const float*)d_in[10];
  const float* wo = (const float*)d_in[11];
  const float* bo = (const float*)d_in[12];
  const float* w1 = (const float*)d_in[13];
  const float* b1 = (const float*)d_in[14];
  const float* w2 = (const float*)d_in[15];
  const float* b2 = (const float*)d_in[16];
  float* out = (float*)d_out;

  const int Bz = 4, S = 1024, Dm = 1024, H = 16, Dff = 4096;
  const int M = Bz * S;  // 4096 token rows
  const size_t TD = (size_t)M * Dm;
  const size_t TF = (size_t)M * Dff;
  const size_t DD = (size_t)Dm * Dm;
  const size_t DF = (size_t)Dm * Dff;

  // workspace carve-up
  float* x1   = (float*)d_ws;           // TD fp32
  __bf16* h1  = (__bf16*)(x1 + TD);     // bf16 from here on
  __bf16* q   = h1 + TD;
  __bf16* k   = q + TD;
  __bf16* v   = k + TD;
  __bf16* att = v + TD;
  __bf16* h2  = att + TD;
  __bf16* f1  = h2 + TD;                // TF
  __bf16* wqT = f1 + TF;                // DD each
  __bf16* wkT = wqT + DD;
  __bf16* wvT = wkT + DD;
  __bf16* woT = wvT + DD;
  __bf16* w1T = woT + DD;               // DF
  __bf16* w2T = w1T + DF;               // DF

  dim3 blk(256);

  cvt_transpose_kernel<<<DD / 256, blk, 0, stream>>>(wq, wqT, Dm, Dm);
  cvt_transpose_kernel<<<DD / 256, blk, 0, stream>>>(wk, wkT, Dm, Dm);
  cvt_transpose_kernel<<<DD / 256, blk, 0, stream>>>(wv, wvT, Dm, Dm);
  cvt_transpose_kernel<<<DD / 256, blk, 0, stream>>>(wo, woT, Dm, Dm);
  cvt_transpose_kernel<<<DF / 256, blk, 0, stream>>>(w1, w1T, Dm, Dff);
  cvt_transpose_kernel<<<DF / 256, blk, 0, stream>>>(w2, w2T, Dff, Dm);

  dim3 gD(Dm / 128, M / 128);
  dim3 gF(Dff / 128, M / 128);
  dim3 gA(Bz * H, S / 128);

  layernorm_kernel<<<M, blk, 0, stream>>>(x, ln1g, ln1b, h1);
  gemm_bf16_kernel<false, false, __bf16><<<gD, blk, 0, stream>>>(h1, wqT, bq, nullptr, q, M, Dm, Dm);
  gemm_bf16_kernel<false, false, __bf16><<<gD, blk, 0, stream>>>(h1, wkT, bk, nullptr, k, M, Dm, Dm);
  gemm_bf16_kernel<false, false, __bf16><<<gD, blk, 0, stream>>>(h1, wvT, bv, nullptr, v, M, Dm, Dm);
  flash_attn_kernel<<<gA, blk, 0, stream>>>(q, k, v, att, H, S, Dm);
  gemm_bf16_kernel<false, true, float><<<gD, blk, 0, stream>>>(att, woT, bo, x, x1, M, Dm, Dm);
  layernorm_kernel<<<M, blk, 0, stream>>>(x1, ln2g, ln2b, h2);
  gemm_bf16_kernel<true, false, __bf16><<<gF, blk, 0, stream>>>(h2, w1T, b1, nullptr, f1, M, Dff, Dm);
  gemm_bf16_kernel<false, true, float><<<gD, blk, 0, stream>>>(f1, w2T, b2, x1, out, M, Dm, Dff);
}
